// GCNModel_31903017074708
// MI455X (gfx1250) — matrix operand order repaired and checked by reading, not compiled, not run on hardware
//
#include <hip/hip_runtime.h>

#define EMB 128
#define N_LIG 32768
#define N_REC 262144
#define NUM_GRAPHS 1024
#define E_LIG 327680
#define E_REC 2097152
#define KREC 199
#define KREC_PAD 224
#define KREC_STRIDE 256
#define BN_EPS 1e-5f

typedef __attribute__((ext_vector_type(16))) __bf16 v16bf;
typedef __attribute__((ext_vector_type(8)))  float v8f;

__device__ __forceinline__ v8f wmma_bf16(v16bf a, v16bf b, v8f c) {
  // (neg_a, A, neg_b, B, c_mod, C, reuse_a, reuse_b)
  return __builtin_amdgcn_wmma_f32_16x16x32_bf16(false, a, false, b, (short)0, c,
                                                 false, false);
}

// ---------------- weight preparation (bf16 + transpose) ----------------

// src [128,128] row-major -> dst [n][k] (transposed), bf16
__global__ void cast_T128_kernel(const float* __restrict__ src,
                                 __bf16* __restrict__ dst) {
  int i = blockIdx.x * blockDim.x + threadIdx.x;  // 128*128 threads
  int n = i >> 7, k = i & 127;
  dst[n * EMB + k] = (__bf16)src[k * EMB + n];
}

// rec_w [199,128] -> dst [128][256] transposed, zero-padded k>=199, bf16
__global__ void cast_T_recw_kernel(const float* __restrict__ src,
                                   __bf16* __restrict__ dst) {
  int i = blockIdx.x * blockDim.x + threadIdx.x;  // 128*256 threads
  int n = i >> 8, k = i & 255;
  float v = (k < KREC) ? src[k * EMB + n] : 0.0f;
  dst[n * KREC_STRIDE + k] = (__bf16)v;
}

// ---------------- utility kernels ----------------

__global__ void fill_kernel(float* __restrict__ p, float v, int n) {
  int i = blockIdx.x * blockDim.x + threadIdx.x;
  if (i < n) p[i] = v;
}

__global__ void zero_kernel(float* __restrict__ p, int n) {
  int i = blockIdx.x * blockDim.x + threadIdx.x;
  if (i < n) p[i] = 0.0f;
}

__global__ void rsqrt_kernel(float* __restrict__ p, int n) {
  int i = blockIdx.x * blockDim.x + threadIdx.x;
  if (i < n) p[i] = rsqrtf(p[i]);
}

__global__ void deg_count_kernel(const int* __restrict__ dst_idx,
                                 float* __restrict__ deg) {
  int i = blockIdx.x * blockDim.x + threadIdx.x;
  atomicAdd(&deg[dst_idx[i]], 1.0f);
}

// ---------------- encoders ----------------

// lig_h[n][c] = sum_f atom_emb[lig_x[n][f]+off[f]][c]
__global__ void lig_encode_kernel(const int* __restrict__ lig_x,
                                  const int* __restrict__ offs,
                                  const float* __restrict__ emb,
                                  float* __restrict__ out) {
  int tid = blockIdx.x * blockDim.x + threadIdx.x;
  int n = tid >> 7, c = tid & 127;
  float s = 0.0f;
#pragma unroll
  for (int f = 0; f < 9; ++f) {
    int idx = lig_x[n * 9 + f] + offs[f];
    s += emb[(size_t)idx * EMB + c];
  }
  out[(size_t)n * EMB + c] = s;
}

// rec_h = rec_emb1[rec_x_int] + rec_x_float @ rec_w + rec_b
// One wave per 16-row slab; 8 accumulators cover all 128 output columns.
__global__ void rec_encode_wmma_kernel(const float* __restrict__ Xf,
                                       const __bf16* __restrict__ Wt,
                                       const int* __restrict__ xi,
                                       const float* __restrict__ emb1,
                                       const float* __restrict__ bias,
                                       float* __restrict__ Y) {
  int tm = ((blockIdx.x * blockDim.x) + threadIdx.x) >> 5;  // slab index
  int lane = threadIdx.x & 31;
  int r = tm * 16 + (lane & 15);
  int khalf = (lane < 16) ? 0 : 16;
  int koff = (lane < 16) ? 0 : 16;
  int colb = lane & 15;
  const float* xrow = Xf + (size_t)r * KREC;

  v8f acc[8] = {};
  for (int k0 = 0; k0 < KREC_PAD; k0 += 32) {
    v16bf a;
#pragma unroll
    for (int e = 0; e < 8; ++e) {
      int k = k0 + khalf + e;
      a[e] = (k < KREC) ? (__bf16)xrow[k] : (__bf16)0.0f;
    }
#pragma unroll
    for (int e = 0; e < 8; ++e) {
      int k = k0 + khalf + 8 + e;
      a[8 + e] = (k < KREC) ? (__bf16)xrow[k] : (__bf16)0.0f;
    }
#pragma unroll
    for (int tn = 0; tn < 8; ++tn) {
      // 32 contiguous bytes, 32B-aligned -> paired b128 loads
      v16bf b = *(const v16bf*)(Wt + (size_t)(tn * 16 + colb) * KREC_STRIDE +
                                k0 + koff);
      acc[tn] = wmma_bf16(a, b, acc[tn]);
    }
  }
  int roff = (lane < 16) ? 0 : 8;
#pragma unroll
  for (int tn = 0; tn < 8; ++tn) {
    int col = tn * 16 + colb;
#pragma unroll
    for (int j = 0; j < 8; ++j) {
      int row = tm * 16 + j + roff;
      Y[(size_t)row * EMB + col] =
          acc[tn][j] + emb1[(size_t)xi[row] * EMB + col] + bias[col];
    }
  }
}

// ---------------- GCN dense transform: Y = X @ W (M x 128 x 128) ----------------
// Wt is the bf16 weight stored transposed [n][k]. One wave per 16-row slab.

__global__ void xw_wmma_kernel(const float* __restrict__ X,
                               const __bf16* __restrict__ Wt,
                               float* __restrict__ Y) {
  int tm = ((blockIdx.x * blockDim.x) + threadIdx.x) >> 5;
  int lane = threadIdx.x & 31;
  int r = tm * 16 + (lane & 15);
  int khalf = (lane < 16) ? 0 : 16;
  int koff = (lane < 16) ? 0 : 16;
  int colb = lane & 15;
  const float* xrow = X + (size_t)r * EMB;

  v8f acc[8] = {};
#pragma unroll
  for (int k0 = 0; k0 < EMB; k0 += 32) {
    const float* ap = xrow + k0 + khalf;
    float4 p0 = *(const float4*)(ap);
    float4 p1 = *(const float4*)(ap + 4);
    float4 p2 = *(const float4*)(ap + 8);
    float4 p3 = *(const float4*)(ap + 12);
    v16bf a;
    a[0] = (__bf16)p0.x;  a[1] = (__bf16)p0.y;  a[2] = (__bf16)p0.z;  a[3] = (__bf16)p0.w;
    a[4] = (__bf16)p1.x;  a[5] = (__bf16)p1.y;  a[6] = (__bf16)p1.z;  a[7] = (__bf16)p1.w;
    a[8] = (__bf16)p2.x;  a[9] = (__bf16)p2.y;  a[10] = (__bf16)p2.z; a[11] = (__bf16)p2.w;
    a[12] = (__bf16)p3.x; a[13] = (__bf16)p3.y; a[14] = (__bf16)p3.z; a[15] = (__bf16)p3.w;
#pragma unroll
    for (int tn = 0; tn < 8; ++tn) {
      v16bf b = *(const v16bf*)(Wt + (size_t)(tn * 16 + colb) * EMB + k0 + koff);
      acc[tn] = wmma_bf16(a, b, acc[tn]);
    }
  }
  int roff = (lane < 16) ? 0 : 8;
#pragma unroll
  for (int tn = 0; tn < 8; ++tn) {
    int col = tn * 16 + colb;
#pragma unroll
    for (int j = 0; j < 8; ++j)
      Y[(size_t)(tm * 16 + j + roff) * EMB + col] = acc[tn][j];
  }
}

// ---------------- message passing ----------------

// out[n][c] = hw[n][c] * dinv[n]^2   (self loop term; also inits accumulator)
__global__ void self_init_kernel(float* __restrict__ out,
                                 const float* __restrict__ hw,
                                 const float* __restrict__ dinv) {
  int i = blockIdx.x * blockDim.x + threadIdx.x;
  int n = i >> 7;
  float d = dinv[n];
  out[i] = hw[i] * d * d;
}

// out[dst] += hw[src] * dinv[src]*dinv[dst]   (4 channels per thread)
__global__ void edge_agg_kernel(const int* __restrict__ src,
                                const int* __restrict__ dst,
                                const float* __restrict__ hw,
                                const float* __restrict__ dinv,
                                float* __restrict__ out) {
  int tid = blockIdx.x * blockDim.x + threadIdx.x;
  int e = tid >> 5;
  int c0 = (tid & 31) * 4;
  int s = src[e], d = dst[e];
  float w = dinv[s] * dinv[d];
  const float* hs = hw + (size_t)s * EMB + c0;
  float* od = out + (size_t)d * EMB + c0;
  atomicAdd(od + 0, hs[0] * w);
  atomicAdd(od + 1, hs[1] * w);
  atomicAdd(od + 2, hs[2] * w);
  atomicAdd(od + 3, hs[3] * w);
}

// ---------------- batch-norm ----------------

__global__ void bn_partial_kernel(const float* __restrict__ X,
                                  float* __restrict__ stats, int N) {
  int c = threadIdx.x;  // 128 threads
  float s = 0.0f, q = 0.0f;
  for (int n = blockIdx.x; n < N; n += gridDim.x) {
    float v = X[(size_t)n * EMB + c];
    s += v;
    q += v * v;
  }
  atomicAdd(&stats[c], s);
  atomicAdd(&stats[EMB + c], q);
}

// x = relu((x - mu) * rsqrt(var+eps) * gamma + beta)   (GCN bias cancels in BN)
__global__ void bn_apply_kernel(float* __restrict__ X,
                                const float* __restrict__ stats,
                                const float* __restrict__ gamma,
                                const float* __restrict__ beta, int N) {
  int i = blockIdx.x * blockDim.x + threadIdx.x;
  int c = i & 127;
  float inv_n = 1.0f / (float)N;
  float mu = stats[c] * inv_n;
  float var = stats[EMB + c] * inv_n - mu * mu;
  float y = (X[i] - mu) * rsqrtf(var + BN_EPS) * gamma[c] + beta[c];
  X[i] = fmaxf(y, 0.0f);
}

// ---------------- readout ----------------

// contiguous batch: graph g owns nodes [g*per, (g+1)*per)
__global__ void graph_mean_kernel(const float* __restrict__ h,
                                  float* __restrict__ feat, int per) {
  int g = blockIdx.x, c = threadIdx.x;
  const float* base = h + (size_t)g * per * EMB;
  float s = 0.0f;
  for (int i = 0; i < per; ++i) s += base[(size_t)i * EMB + c];
  feat[(size_t)g * EMB + c] = s / (float)per;
}

__global__ void head_kernel(const float* __restrict__ ligf,
                            const float* __restrict__ recf,
                            const float* __restrict__ w1,
                            const float* __restrict__ b1,
                            const float* __restrict__ w2,
                            const float* __restrict__ b2,
                            float* __restrict__ out) {
  int g = blockIdx.x, t = threadIdx.x;  // 128 threads
  const float* zl = ligf + (size_t)g * EMB;
  const float* zr = recf + (size_t)g * EMB;
  float acc = b1[t];
  for (int k = 0; k < EMB; ++k) acc += zl[k] * w1[(size_t)k * EMB + t];
  for (int k = 0; k < EMB; ++k) acc += zr[k] * w1[(size_t)(EMB + k) * EMB + t];
  acc = fmaxf(acc, 0.0f) * w2[t];
  __shared__ float sm[EMB];
  sm[t] = acc;
  __syncthreads();
  for (int s = 64; s > 0; s >>= 1) {
    if (t < s) sm[t] += sm[t + s];
    __syncthreads();
  }
  if (t == 0) out[g] = sm[0] + b2[0];
}

// ---------------- host ----------------

extern "C" void kernel_launch(void* const* d_in, const int* in_sizes, int n_in,
                              void* d_out, int out_size, void* d_ws,
                              size_t ws_size, hipStream_t stream) {
  (void)in_sizes; (void)n_in; (void)out_size; (void)ws_size;

  const int*   lig_x     = (const int*)d_in[0];
  const int*   rec_xi    = (const int*)d_in[1];
  const float* rec_xf    = (const float*)d_in[2];
  const int*   lig_ei    = (const int*)d_in[3];
  const int*   rec_ei    = (const int*)d_in[4];
  const int*   atom_off  = (const int*)d_in[7];
  const float* atom_emb  = (const float*)d_in[8];
  const float* rec_emb1  = (const float*)d_in[9];
  const float* rec_w     = (const float*)d_in[10];
  const float* rec_b     = (const float*)d_in[11];
  const float* lig_W     = (const float*)d_in[12];
  const float* lig_gamma = (const float*)d_in[14];
  const float* lig_beta  = (const float*)d_in[15];
  const float* rec_W     = (const float*)d_in[16];
  const float* rec_gamma = (const float*)d_in[18];
  const float* rec_beta  = (const float*)d_in[19];
  const float* out_w1    = (const float*)d_in[20];
  const float* out_b1    = (const float*)d_in[21];
  const float* out_w2    = (const float*)d_in[22];
  const float* out_b2    = (const float*)d_in[23];
  float* out = (float*)d_out;

  char* ws = (char*)d_ws;
  size_t off = 0;
  auto take = [&](size_t bytes) -> char* {
    char* p = ws + off;
    off += (bytes + 255) & ~(size_t)255;
    return p;
  };
  float*  recA     = (float*)take((size_t)N_REC * EMB * 4);
  float*  recB     = (float*)take((size_t)N_REC * EMB * 4);
  float*  ligA     = (float*)take((size_t)N_LIG * EMB * 4);
  float*  ligB     = (float*)take((size_t)N_LIG * EMB * 4);
  float*  rec_dinv = (float*)take((size_t)N_REC * 4);
  float*  lig_dinv = (float*)take((size_t)N_LIG * 4);
  __bf16* wrec_enc = (__bf16*)take((size_t)EMB * KREC_STRIDE * 2);
  __bf16* wligW    = (__bf16*)take((size_t)3 * EMB * EMB * 2);
  __bf16* wrecW    = (__bf16*)take((size_t)3 * EMB * EMB * 2);
  float*  stats    = (float*)take(2 * EMB * 4);
  float*  ligfeat  = (float*)take((size_t)NUM_GRAPHS * EMB * 4);
  float*  recfeat  = (float*)take((size_t)NUM_GRAPHS * EMB * 4);

  // weight preparation: bf16 + transpose (encoder K padded to stride 256)
  cast_T_recw_kernel<<<(EMB * KREC_STRIDE) / 256, 256, 0, stream>>>(rec_w, wrec_enc);
  for (int l = 0; l < 3; ++l) {
    cast_T128_kernel<<<(EMB * EMB) / 256, 256, 0, stream>>>(
        lig_W + (size_t)l * EMB * EMB, wligW + (size_t)l * EMB * EMB);
    cast_T128_kernel<<<(EMB * EMB) / 256, 256, 0, stream>>>(
        rec_W + (size_t)l * EMB * EMB, wrecW + (size_t)l * EMB * EMB);
  }

  // encoders
  lig_encode_kernel<<<(N_LIG * EMB) / 256, 256, 0, stream>>>(lig_x, atom_off, atom_emb, ligA);
  rec_encode_wmma_kernel<<<N_REC / 128, 256, 0, stream>>>(rec_xf, wrec_enc, rec_xi,
                                                          rec_emb1, rec_b, recA);

  // degrees -> dinv (self loops included via init=1)
  fill_kernel<<<N_LIG / 256, 256, 0, stream>>>(lig_dinv, 1.0f, N_LIG);
  deg_count_kernel<<<E_LIG / 256, 256, 0, stream>>>(lig_ei + E_LIG, lig_dinv);
  rsqrt_kernel<<<N_LIG / 256, 256, 0, stream>>>(lig_dinv, N_LIG);
  fill_kernel<<<N_REC / 256, 256, 0, stream>>>(rec_dinv, 1.0f, N_REC);
  deg_count_kernel<<<E_REC / 256, 256, 0, stream>>>(rec_ei + E_REC, rec_dinv);
  rsqrt_kernel<<<N_REC / 256, 256, 0, stream>>>(rec_dinv, N_REC);

  for (int l = 0; l < 3; ++l) {
    // ---- ligand side ----
    xw_wmma_kernel<<<N_LIG / 128, 256, 0, stream>>>(ligA, wligW + (size_t)l * EMB * EMB, ligB);
    self_init_kernel<<<(N_LIG * EMB) / 256, 256, 0, stream>>>(ligA, ligB, lig_dinv);
    edge_agg_kernel<<<((size_t)E_LIG * 32) / 256, 256, 0, stream>>>(
        lig_ei, lig_ei + E_LIG, ligB, lig_dinv, ligA);
    zero_kernel<<<1, 256, 0, stream>>>(stats, 2 * EMB);
    bn_partial_kernel<<<512, 128, 0, stream>>>(ligA, stats, N_LIG);
    bn_apply_kernel<<<(N_LIG * EMB) / 256, 256, 0, stream>>>(
        ligA, stats, lig_gamma + l * EMB, lig_beta + l * EMB, N_LIG);
    // ---- receptor side ----
    xw_wmma_kernel<<<N_REC / 128, 256, 0, stream>>>(recA, wrecW + (size_t)l * EMB * EMB, recB);
    self_init_kernel<<<(N_REC * EMB) / 256, 256, 0, stream>>>(recA, recB, rec_dinv);
    edge_agg_kernel<<<((size_t)E_REC * 32) / 256, 256, 0, stream>>>(
        rec_ei, rec_ei + E_REC, recB, rec_dinv, recA);
    zero_kernel<<<1, 256, 0, stream>>>(stats, 2 * EMB);
    bn_partial_kernel<<<512, 128, 0, stream>>>(recA, stats, N_REC);
    bn_apply_kernel<<<(N_REC * EMB) / 256, 256, 0, stream>>>(
        recA, stats, rec_gamma + l * EMB, rec_beta + l * EMB, N_REC);
  }

  // pooling (contiguous batches) + head MLP
  graph_mean_kernel<<<NUM_GRAPHS, 128, 0, stream>>>(ligA, ligfeat, N_LIG / NUM_GRAPHS);
  graph_mean_kernel<<<NUM_GRAPHS, 128, 0, stream>>>(recA, recfeat, N_REC / NUM_GRAPHS);
  head_kernel<<<NUM_GRAPHS, 128, 0, stream>>>(ligfeat, recfeat, out_w1, out_b1,
                                              out_w2, out_b2, out);
}
